// SelfAttention_22720376996032
// MI455X (gfx1250) — matrix operand order repaired and checked by reading, not compiled, not run on hardware
//
#include <hip/hip_runtime.h>

typedef __attribute__((ext_vector_type(16))) _Float16 v16h;
typedef __attribute__((ext_vector_type(8)))  float    v8f;
typedef __attribute__((ext_vector_type(4)))  unsigned v4u;
typedef __attribute__((ext_vector_type(8)))  int      v8i;
typedef __attribute__((ext_vector_type(4)))  int      v4i;

#define N_BATCH 2
#define SEQ     2048
#define EMBED   1024
#define HEADS   16
#define HDIM    64

#define KT_PAD  72   // K tile row stride in halfs (64 data + 8 pad = 36 dwords)
#define VT_PAD  36   // V tile row stride in halfs (32 data + 4 pad = 18 dwords)

// D = A(16x32 f16) * B(32x16 f16) + C(16x16 f32)
__device__ inline v8f wmma_f16(v16h a, v16h b, v8f c) {
    return __builtin_amdgcn_wmma_f32_16x16x32_f16(
        /*neg_a=*/false, a, /*neg_b=*/false, b,
        /*c_mod=*/(short)0, c, /*reuse_a=*/false, /*reuse_b=*/false);
}

__device__ inline unsigned lds_off_of(const void* p) {
    return (unsigned)(size_t)p;   // low 32 bits of shared aperture = LDS offset
}

// ---- Tensor Data Mover: D# group 0 (count=1, lds_addr, global_addr, type=2)
__device__ inline v4u tdm_g0(unsigned lds_byte_off, const void* gaddr) {
    const unsigned long long ga = (unsigned long long)gaddr;
    v4u g;
    g.x = 1u;                                                  // count=1, user
    g.y = lds_byte_off;                                        // bits 63:32
    g.z = (unsigned)ga;                                        // bits 95:64
    g.w = ((unsigned)(ga >> 32) & 0x01FFFFFFu) | 0x80000000u;  // ga[56:32]|type=2
    return g;
}

// ---- D# group 1: data_size=2B, LDS pad enabled, 2D tensor/tile, stride0.
// d0 bits: [17:16]=data_size(1->2B) [20]=pad_en [24:22]=pad_interval [31:25]=pad_amount
__device__ inline v8i tdm_g1(unsigned pad_interval, unsigned pad_amount,
                             unsigned dim0, unsigned dim1,
                             unsigned tile0, unsigned tile1, unsigned stride0) {
    v8i g = {};
    g[0] = (int)((1u << 16) | (1u << 20) | (pad_interval << 22) | (pad_amount << 25));
    g[1] = (int)(dim0 << 16);                        // tensor_dim0[15:0] @ bits 63:48
    g[2] = (int)((dim0 >> 16) | (dim1 << 16));       // dim0[31:16], dim1[15:0]
    g[3] = (int)((dim1 >> 16) | (tile0 << 16));      // dim1[31:16], tile_dim0
    g[4] = (int)tile1;                               // tile_dim1 (tile_dim2=0 -> 2D)
    g[5] = (int)stride0;                             // tensor_dim0_stride[31:0]
    return g;
}

__device__ inline void tdm_load(v4u g0, v8i g1) {
#if __has_builtin(__builtin_amdgcn_tensor_load_to_lds)
    v4i z4 = {};
    v8i z8 = {};
    __builtin_amdgcn_tensor_load_to_lds(g0, g1, z4, z4, z8, 0);
#else
    asm volatile("tensor_load_to_lds %0, %1" :: "s"(g0), "s"(g1) : "memory");
#endif
}

// A-matrix fragment (16x32, MxK), row-major source with row stride `ld` elems.
// lane m holds row m; K 0..15 (lanes 0-15) / 16..31 (lanes 16-31), matching load_bfrag.
template <typename T>
__device__ inline v16h load_afrag(const T* __restrict__ p, int ld, int lane) {
    const int m    = lane & 15;
    const int koff = (lane & 16) ? 16 : 0;
    const T* r = p + (size_t)m * ld;
    v16h a;
#pragma unroll
    for (int i = 0; i < 8; ++i) {
        const int k = 2 * i + koff;
        a[2 * i]     = (_Float16)r[k];
        a[2 * i + 1] = (_Float16)r[k + 1];
    }
    return a;
}

// B-matrix fragment (32x16, KxN). B[k][n] at p + n*col_stride + k (contiguous k).
// Contiguous k order (not the ISA table order): lanes 0-15 hold col n, K=0..15; lanes 16-31 hold K=16..31.
template <typename T>
__device__ inline v16h load_bfrag(const T* __restrict__ p, int col_stride, int lane) {
    const int n    = lane & 15;
    const int koff = (lane & 16) ? 16 : 0;
    const T* c = p + (size_t)n * col_stride + koff;
    v16h b;
#pragma unroll
    for (int i = 0; i < 16; ++i) b[i] = (_Float16)c[i];
    return b;
}

// ---------------------------------------------------------------------------
// Kernel 1: per-head QKV projections.  y[n,s,h,e] = sum_d x[n,s,h,d]*W[e,d]
// Q', K' written (N,H,S,D) f16;  V' written transposed (N,H,D,S) f16.
// grid = (SEQ/16, N*H, 3), block = 32.
// ---------------------------------------------------------------------------
__global__ void qkv_proj_kernel(const float* __restrict__ q,
                                const float* __restrict__ k,
                                const float* __restrict__ v,
                                const float* __restrict__ Wq,
                                const float* __restrict__ Wk,
                                const float* __restrict__ Wv,
                                _Float16* __restrict__ Qp,
                                _Float16* __restrict__ Kp,
                                _Float16* __restrict__ Vp) {
    const int lane = threadIdx.x & 31;
    const int s0   = blockIdx.x * 16;
    const int nh   = blockIdx.y;
    const int n    = nh >> 4;
    const int h    = nh & 15;

    const float* X; const float* W; _Float16* Y;
    if (blockIdx.z == 0)      { X = q; W = Wq; Y = Qp; }
    else if (blockIdx.z == 1) { X = k; W = Wk; Y = Kp; }
    else                      { X = v; W = Wv; Y = Vp; }

    const float* abase = X + ((size_t)(n * SEQ + s0)) * EMBED + h * HDIM;

    v8f acc[4] = {};
#pragma unroll
    for (int k0 = 0; k0 < HDIM; k0 += 32) {
        v16h a = load_afrag(abase + k0, EMBED, lane);
#pragma unroll
        for (int t = 0; t < 4; ++t) {
            // B[k][e] = W[(t*16+e)][k0+k]
            v16h b = load_bfrag(W + (size_t)(t * 16) * HDIM + k0, HDIM, lane);
            acc[t] = wmma_f16(a, b, acc[t]);
        }
    }

    const int colb = lane & 15;
    const int rowb = (lane >> 4) * 8;
    const size_t hoff = (size_t)(n * HEADS + h);
#pragma unroll
    for (int t = 0; t < 4; ++t)
#pragma unroll
        for (int vi = 0; vi < 8; ++vi) {
            const int s = s0 + rowb + vi;
            const int e = t * 16 + colb;
            if (blockIdx.z == 2)   // V transposed: (N,H,D,S)
                Y[(hoff * HDIM + e) * SEQ + s] = (_Float16)acc[t][vi];
            else                   // Q,K: (N,H,S,D)
                Y[(hoff * SEQ + s) * HDIM + e] = (_Float16)acc[t][vi];
        }
}

// ---------------------------------------------------------------------------
// Kernel 2: fused flash attention. Block = 4 waves, one (n,h), 64 q rows.
// K/V tiles (32 keys) double-buffered in LDS via the Tensor Data Mover
// (wave 0 issues 2 TDM descriptors/tile; TENSORcnt + barrier publishes).
// grid = (SEQ/64, N*H), block = 128.
// ---------------------------------------------------------------------------
__global__ void attn_kernel(const _Float16* __restrict__ Qp,
                            const _Float16* __restrict__ Kp,   // (N,H,S,D)
                            const _Float16* __restrict__ Vp,   // (N,H,D,S)
                            const int* __restrict__ mask,
                            _Float16* __restrict__ Aout) {
    __shared__ _Float16 kt[2][32][KT_PAD];   // (key, d), TDM-padded rows
    __shared__ _Float16 vt[2][HDIM][VT_PAD]; // (d, key), TDM-padded rows
    __shared__ _Float16 pbuf[4][16][34];     // per-wave P staging

    const int tid  = threadIdx.x;
    const int lane = tid & 31;
    const int w    = tid >> 5;
    const int nh   = blockIdx.y;
    const int n    = nh >> 4;
    const int h    = nh & 15;
    const int q0   = (blockIdx.x * 4 + w) * 16;

    const size_t head_off = ((size_t)(n * HEADS + h));
    const _Float16* kg0 = Kp + (head_off * SEQ) * HDIM;   // + k0*HDIM
    const _Float16* vg0 = Vp + (head_off * HDIM) * SEQ;   // rows d, + k0

    // K tile: 32 rows x 64 halfs (128B = 32 dwords) -> pad_interval=4, amount=3 (4dw)
    const v8i g1k = tdm_g1(4u, 3u, HDIM, SEQ, HDIM, 32u, HDIM);
    // V tile: 64 rows x 32 halfs (64B = 16 dwords)  -> pad_interval=3, amount=1 (2dw)
    const v8i g1v = tdm_g1(3u, 1u, 32u, HDIM, 32u, HDIM, SEQ);

    // Q fragments (K-dim 64 -> two 16x32 frags), loaded once.
    const _Float16* qbase = Qp + (head_off * SEQ + q0) * HDIM;
    const v16h aq0 = load_afrag(qbase,      HDIM, lane);
    const v16h aq1 = load_afrag(qbase + 32, HDIM, lane);

    const int   colb   = lane & 15;
    const int   rowb   = (lane >> 4) * 8;
    const float scale  = 0.03125f;      // 1/sqrt(EMBED)
    const float MASKED = -3.125e18f;    // -1e20 * scale

    float mrow[8], lrow[8];
#pragma unroll
    for (int i = 0; i < 8; ++i) { mrow[i] = -3.0e38f; lrow[i] = 0.0f; }
    v8f o[4] = {};

    // Wave-uniform branch: only wave 0 ever issues TDM (EXEC-independent op).
    auto prefetch = [&](int buf, int k0) {
        tdm_load(tdm_g0(lds_off_of(&kt[buf][0][0]), kg0 + (size_t)k0 * HDIM), g1k);
        tdm_load(tdm_g0(lds_off_of(&vt[buf][0][0]), vg0 + k0),                g1v);
    };

    if (w == 0) prefetch(0, 0);

    for (int k0 = 0; k0 < SEQ; k0 += 32) {
        const int  cur  = (k0 >> 5) & 1;
        const bool more = (k0 + 32) < SEQ;
        if (w == 0) {
            if (more) {
                prefetch(cur ^ 1, k0 + 32);
                __builtin_amdgcn_s_wait_tensorcnt(2);  // retire current tile
            } else {
                __builtin_amdgcn_s_wait_tensorcnt(0);
            }
        }
        __syncthreads();   // current tile visible to all waves

        // ----- S = Q * K^T (two 16-key score tiles) -----
        v8f s0 = {}, s1 = {};
        {
            v16h b0 = load_bfrag(&kt[cur][0][0],   KT_PAD, lane);
            v16h b1 = load_bfrag(&kt[cur][0][32],  KT_PAD, lane);
            s0 = wmma_f16(aq0, b0, s0);
            s0 = wmma_f16(aq1, b1, s0);
            v16h c0 = load_bfrag(&kt[cur][16][0],  KT_PAD, lane);
            v16h c1 = load_bfrag(&kt[cur][16][32], KT_PAD, lane);
            s1 = wmma_f16(aq0, c0, s1);
            s1 = wmma_f16(aq1, c1, s1);
        }

        // ----- mask + scale (C layout: row = rowb+vi, col = colb / 16+colb)
        float sv0[8], sv1[8];
#pragma unroll
        for (int vi = 0; vi < 8; ++vi) {
            const int qrow = q0 + rowb + vi;
            const int* mp  = mask + ((size_t)n * SEQ + qrow) * SEQ + k0;
            sv0[vi] = (mp[colb]      == 0) ? MASKED : s0[vi] * scale;
            sv1[vi] = (mp[16 + colb] == 0) ? MASKED : s1[vi] * scale;
        }

        // ----- online softmax -----
        float newm[8], corr[8];
#pragma unroll
        for (int vi = 0; vi < 8; ++vi) {
            float mx = fmaxf(sv0[vi], sv1[vi]);
#pragma unroll
            for (int off = 8; off >= 1; off >>= 1)
                mx = fmaxf(mx, __shfl_xor(mx, off, 32));
            newm[vi] = fmaxf(mrow[vi], mx);
            corr[vi] = __expf(mrow[vi] - newm[vi]);
            mrow[vi] = newm[vi];
        }
#pragma unroll
        for (int vi = 0; vi < 8; ++vi) {
            const float p0 = __expf(sv0[vi] - newm[vi]);
            const float p1 = __expf(sv1[vi] - newm[vi]);
            float rs = p0 + p1;
#pragma unroll
            for (int off = 8; off >= 1; off >>= 1)
                rs += __shfl_xor(rs, off, 32);
            lrow[vi] = lrow[vi] * corr[vi] + rs;
            pbuf[w][rowb + vi][colb]      = (_Float16)p0;
            pbuf[w][rowb + vi][16 + colb] = (_Float16)p1;
        }

        asm volatile("s_wait_dscnt 0x0" ::: "memory");  // same-wave P store->load
        const v16h pa = load_afrag(&pbuf[w][0][0], 34, lane);

        // ----- O = diag(corr)*O + P * V  (V tile is (d,key) in LDS) -----
#pragma unroll
        for (int t = 0; t < 4; ++t) {
#pragma unroll
            for (int vi = 0; vi < 8; ++vi) o[t][vi] *= corr[vi];
            v16h bv = load_bfrag(&vt[cur][t * 16][0], VT_PAD, lane);
            o[t] = wmma_f16(pa, bv, o[t]);
        }

        __syncthreads();   // all waves done with `cur` before it is refilled
    }

    // ----- normalize, store attention output (N, S, EMBED) f16 -----
#pragma unroll
    for (int t = 0; t < 4; ++t)
#pragma unroll
        for (int vi = 0; vi < 8; ++vi) {
            const int   qrow = q0 + rowb + vi;
            const float val  = o[t][vi] / lrow[vi];
            Aout[((size_t)(n * SEQ + qrow)) * EMBED + h * HDIM + t * 16 + colb] =
                (_Float16)val;
        }
}

// ---------------------------------------------------------------------------
// Wo fp32 -> f16 one-shot conversion. grid = (EMBED*EMBED/1024), block = 256.
// ---------------------------------------------------------------------------
__global__ void f32_to_f16_kernel(const float* __restrict__ src,
                                  _Float16* __restrict__ dst) {
    const int i = (blockIdx.x * blockDim.x + threadIdx.x) * 4;
    const float4 f = *(const float4*)(src + i);
    dst[i + 0] = (_Float16)f.x;
    dst[i + 1] = (_Float16)f.y;
    dst[i + 2] = (_Float16)f.z;
    dst[i + 3] = (_Float16)f.w;
}

// ---------------------------------------------------------------------------
// Kernel 3: output projection, 64x64 tile per wave (4x4 WMMA C tiles).
// out = A(4096x1024) @ Wo^T + bo.  grid = (M/64, EMBED/64), block = 32.
// ---------------------------------------------------------------------------
__global__ void out_proj_kernel(const _Float16* __restrict__ A,
                                const _Float16* __restrict__ Wo16,
                                const float* __restrict__ bo,
                                float* __restrict__ out) {
    const int lane = threadIdx.x & 31;
    const int m0   = blockIdx.x * 64;
    const int c0   = blockIdx.y * 64;

    v8f acc[4][4] = {};
    for (int k0 = 0; k0 < EMBED; k0 += 32) {
        v16h a[4], b[4];
#pragma unroll
        for (int mt = 0; mt < 4; ++mt)
            a[mt] = load_afrag(A + (size_t)(m0 + mt * 16) * EMBED + k0, EMBED, lane);
#pragma unroll
        for (int t = 0; t < 4; ++t)
            b[t] = load_bfrag(Wo16 + (size_t)(c0 + t * 16) * EMBED + k0, EMBED, lane);
#pragma unroll
        for (int mt = 0; mt < 4; ++mt)
#pragma unroll
            for (int t = 0; t < 4; ++t)
                acc[mt][t] = wmma_f16(a[mt], b[t], acc[mt][t]);
    }

    const int colb = lane & 15;
    const int rowb = (lane >> 4) * 8;
#pragma unroll
    for (int mt = 0; mt < 4; ++mt)
#pragma unroll
        for (int t = 0; t < 4; ++t)
#pragma unroll
            for (int vi = 0; vi < 8; ++vi) {
                const int col = c0 + t * 16 + colb;
                out[((size_t)(m0 + mt * 16 + rowb + vi)) * EMBED + col] =
                    acc[mt][t][vi] + bo[col];
            }
}

extern "C" void kernel_launch(void* const* d_in, const int* in_sizes, int n_in,
                              void* d_out, int out_size, void* d_ws, size_t ws_size,
                              hipStream_t stream) {
    const float* q    = (const float*)d_in[0];
    const float* k    = (const float*)d_in[1];
    const float* v    = (const float*)d_in[2];
    const int*   mask = (const int*)d_in[3];
    const float* Wq   = (const float*)d_in[4];
    const float* Wk   = (const float*)d_in[5];
    const float* Wv   = (const float*)d_in[6];
    const float* Wo   = (const float*)d_in[7];
    const float* bo   = (const float*)d_in[8];
    float*       out  = (float*)d_out;

    const size_t per_buf = (size_t)N_BATCH * HEADS * SEQ * HDIM;  // 4M halfs
    _Float16* Qp   = (_Float16*)d_ws;
    _Float16* Kp   = Qp + per_buf;
    _Float16* Vp   = Kp + per_buf;                 // (N,H,D,S)
    _Float16* Aout = Vp + per_buf;                 // (N,S,EMBED), 4M halfs
    _Float16* Wo16 = Aout + per_buf;               // 1M halfs

    qkv_proj_kernel<<<dim3(SEQ / 16, N_BATCH * HEADS, 3), 32, 0, stream>>>(
        q, k, v, Wq, Wk, Wv, Qp, Kp, Vp);

    f32_to_f16_kernel<<<dim3((EMBED * EMBED) / 1024), 256, 0, stream>>>(Wo, Wo16);

    attn_kernel<<<dim3(SEQ / 64, N_BATCH * HEADS), 128, 0, stream>>>(
        Qp, Kp, Vp, mask, Aout);

    out_proj_kernel<<<dim3((N_BATCH * SEQ) / 64, EMBED / 64), 32, 0, stream>>>(
        Aout, Wo16, bo, out);
}
